// SelfAttention_25795573580634
// MI455X (gfx1250) — matrix operand order repaired and checked by reading, not compiled, not run on hardware
//
#include <hip/hip_runtime.h>
#include <hip/hip_bf16.h>

#define D_MODEL  4096
#define N_HEADS  32
#define HEAD_DIM 128
#define SEQ      512
#define BATCH    4
#define MAXS     4096

typedef __attribute__((ext_vector_type(16))) __bf16          bf16x16;
typedef __attribute__((ext_vector_type(8)))  float           f32x8;
typedef __attribute__((ext_vector_type(8)))  unsigned short  ushort8;

// packed fp32->bf16 (round-to-nearest-even) : 1 VALU op per 2 elements
__device__ __forceinline__ unsigned bf2pk(float a, float b) {
    unsigned r;
    asm("v_cvt_pk_bf16_f32 %0, %1, %2" : "=v"(r) : "v"(a), "v"(b));
    return r;
}
__device__ __forceinline__ unsigned short bf1(float a) {
    return (unsigned short)bf2pk(a, a);
}

union FragBF {
    ushort8 h[2];
    bf16x16 v;
};

// ---------------------------------------------------------------------------
// Kernel 1: QKV projection GEMM.  out[m][n] = sum_k x[m][k]*w[n][k] + bias[n]
// M=2048, N=4096, K=4096.  blockIdx.z selects {Q,K,V}.
// 128x128 tile per block, K-chunk 64, 8 waves, each wave 32x64 (2x4 WMMA
// tiles x 2 k-subchunks = 16 WMMA / wave / iteration), bf16 WMMA.
// ---------------------------------------------------------------------------
__global__ __launch_bounds__(256)
void qkv_gemm_kernel(const float* __restrict__ x,
                     const float* __restrict__ wq, const float* __restrict__ bq,
                     const float* __restrict__ wk, const float* __restrict__ bk,
                     const float* __restrict__ wv, const float* __restrict__ bv,
                     float* __restrict__ qb, float* __restrict__ kb,
                     float* __restrict__ vb)
{
    __shared__ __align__(32) unsigned short Ash[128 * 64];
    __shared__ __align__(32) unsigned short Bsh[128 * 64];

    const int z = blockIdx.z;
    const float* w    = (z == 0) ? wq : (z == 1) ? wk : wv;
    const float* bias = (z == 0) ? bq : (z == 1) ? bk : bv;
    float*       out  = (z == 0) ? qb : (z == 1) ? kb : vb;

    const int m0   = blockIdx.x * 128;
    const int n0   = blockIdx.y * 128;
    const int tid  = threadIdx.x;
    const int lane = tid & 31;
    const int wave = tid >> 5;
    const int wm   = wave & 3;   // 32-row slice
    const int wn   = wave >> 2;  // 64-col slice

    const f32x8 zero8 = {0.f,0.f,0.f,0.f,0.f,0.f,0.f,0.f};
    f32x8 acc[2][4];
    #pragma unroll
    for (int i = 0; i < 2; i++)
        #pragma unroll
        for (int j = 0; j < 4; j++) acc[i][j] = zero8;

    const int srow = tid >> 1;          // 0..127 (staging row)
    const int skp  = (tid & 1) * 32;    // 0 / 32 (staging k offset)

    for (int k0 = 0; k0 < D_MODEL; k0 += 64) {
        __syncthreads();
        // stage A (x rows) and B (w rows == columns of w^T) as packed bf16
        {
            const float4* sa = (const float4*)(x + (size_t)(m0 + srow) * D_MODEL + k0 + skp);
            const float4* sb = (const float4*)(w + (size_t)(n0 + srow) * D_MODEL + k0 + skp);
            #pragma unroll
            for (int i = 0; i < 8; i++) {
                float4 fa = sa[i];
                float4 fb = sb[i];
                int o = srow * 64 + skp + i * 4;
                uint2 pa; pa.x = bf2pk(fa.x, fa.y); pa.y = bf2pk(fa.z, fa.w);
                uint2 pb; pb.x = bf2pk(fb.x, fb.y); pb.y = bf2pk(fb.z, fb.w);
                *(uint2*)&Ash[o] = pa;
                *(uint2*)&Bsh[o] = pb;
            }
            if (k0 + 64 < D_MODEL) {
                __builtin_prefetch(x + (size_t)(m0 + srow) * D_MODEL + k0 + 64 + skp, 0, 0);
                __builtin_prefetch(w + (size_t)(n0 + srow) * D_MODEL + k0 + 64 + skp, 0, 0);
            }
        }
        __syncthreads();

        const int aoff = (lane < 16) ? 0 : 8;  // A k-halves {0..7,16..23}/{8..15,24..31}
        const int boff = (lane >> 4) * 8;      // B k-halves {0..7,16..23}/{8..15,24..31}
        #pragma unroll
        for (int ks = 0; ks < 64; ks += 32) {
            FragBF a[2], b[4];
            #pragma unroll
            for (int mt = 0; mt < 2; mt++) {
                int base = (wm * 32 + mt * 16 + (lane & 15)) * 64 + ks + aoff;
                a[mt].h[0] = *(const ushort8*)&Ash[base];
                a[mt].h[1] = *(const ushort8*)&Ash[base + 16];
            }
            #pragma unroll
            for (int nt = 0; nt < 4; nt++) {
                int base = (wn * 64 + nt * 16 + (lane & 15)) * 64 + ks + boff;
                b[nt].h[0] = *(const ushort8*)&Bsh[base];
                b[nt].h[1] = *(const ushort8*)&Bsh[base + 16];
            }
            #pragma unroll
            for (int mt = 0; mt < 2; mt++)
                #pragma unroll
                for (int nt = 0; nt < 4; nt++)
                    acc[mt][nt] = __builtin_amdgcn_wmma_f32_16x16x32_bf16(
                        false, a[mt].v, false, b[nt].v, (short)0, acc[mt][nt],
                        false, false);
        }
    }

    // epilogue: + bias, store fp32
    #pragma unroll
    for (int nt = 0; nt < 4; nt++) {
        const int n  = n0 + wn * 64 + nt * 16 + (lane & 15);
        const float bs = bias[n];
        #pragma unroll
        for (int mt = 0; mt < 2; mt++) {
            const int rbase = m0 + wm * 32 + mt * 16 + ((lane < 16) ? 0 : 8);
            #pragma unroll
            for (int r = 0; r < 8; r++)
                out[(size_t)(rbase + r) * D_MODEL + n] = acc[mt][nt][r] + bs;
        }
    }
}

// ---------------------------------------------------------------------------
// Kernel 2: RoPE in place on Q and K buffers (blockIdx.y selects buffer).
// ---------------------------------------------------------------------------
__global__ __launch_bounds__(256)
void rope_kernel(float* __restrict__ qb, float* __restrict__ kb,
                 const float* __restrict__ fcos, const float* __restrict__ fsin)
{
    float* buf = (blockIdx.y == 0) ? qb : kb;
    int idx = blockIdx.x * 256 + threadIdx.x;  // pair index, 2048 pairs/row
    int m = idx >> 11;       // row 0..2047  (b*512+s)
    int p = idx & 2047;
    int h = p >> 6;
    int i = p & 63;
    int s = m & (SEQ - 1);
    float c  = fcos[s * 64 + i];
    float sn = fsin[s * 64 + i];
    size_t off = (size_t)m * D_MODEL + h * HEAD_DIM + 2 * i;
    float x0 = buf[off], x1 = buf[off + 1];
    buf[off]     = x0 * c - x1 * sn;
    buf[off + 1] = x0 * sn + x1 * c;
}

// ---------------------------------------------------------------------------
// Kernel 3: flash attention over total = start_pos + SEQ keys.
// Block = 256 thr = 8 waves; block handles (b, h, 128 q-rows); wave owns 16 q.
// ---------------------------------------------------------------------------
__global__ __launch_bounds__(256)
void attn_kernel(const float* __restrict__ qb, const float* __restrict__ kb,
                 const float* __restrict__ vb,
                 const float* __restrict__ cache_k,
                 const float* __restrict__ cache_v,
                 const int* __restrict__ sp, float* __restrict__ out)
{
    __shared__ __align__(32) unsigned short Ksh[32 * 128];  // [key][d]
    __shared__ __align__(32) unsigned short Vt [128 * 32];  // [d][key]
    __shared__ float          Ssh[8 * 16 * 32];             // per-wave scores
    __shared__ __align__(32) unsigned short Psh[8 * 16 * 32];
    __shared__ float          Alp[8 * 16];
    __shared__ float          Sum[8 * 16];

    const int start_pos = sp[0];
    const int total     = start_pos + SEQ;
    const int qblk = blockIdx.x;
    const int h    = blockIdx.y;
    const int b    = blockIdx.z;
    const int tid  = threadIdx.x;
    const int lane = tid & 31;
    const int w    = tid >> 5;

    const float scale = 0.088388347648318447f;  // 1/sqrt(128)

    // Q tile as 4 bf16 A-fragments (K chunks of 32 over head_dim=128)
    FragBF aq[4];
    {
        const int qrow = qblk * 128 + w * 16 + (lane & 15);
        const float* qp = qb + (size_t)(b * SEQ + qrow) * D_MODEL + h * HEAD_DIM;
        const int aoff = (lane < 16) ? 0 : 8;
        #pragma unroll
        for (int c = 0; c < 4; c++) {
            union { unsigned u[8]; ushort8 h8[2]; } u;
            #pragma unroll
            for (int j = 0; j < 4; j++)
                u.u[j]     = bf2pk(qp[c*32 + aoff + 2*j] * scale,
                                   qp[c*32 + aoff + 2*j + 1] * scale);
            #pragma unroll
            for (int j = 0; j < 4; j++)
                u.u[4 + j] = bf2pk(qp[c*32 + aoff + 16 + 2*j] * scale,
                                   qp[c*32 + aoff + 16 + 2*j + 1] * scale);
            aq[c].h[0] = u.h8[0];
            aq[c].h[1] = u.h8[1];
        }
    }

    const f32x8 zero8 = {0.f,0.f,0.f,0.f,0.f,0.f,0.f,0.f};
    f32x8 oacc[8];
    #pragma unroll
    for (int nt = 0; nt < 8; nt++) oacc[nt] = zero8;
    float mrow = -__builtin_inff();
    float srow = 0.f;

    const int key = tid >> 3;         // staging: key 0..31
    const int dp  = (tid & 7) * 16;   // staging: d offset

    for (int kc = 0; kc < total; kc += 32) {
        __syncthreads();
        // ---- stage K (row-major, packed) and V (transposed) as bf16 ----
        {
            const int pos = kc + key;
            const float* krow;
            const float* vrow;
            if (pos < start_pos) {
                size_t co = (size_t)((b * MAXS + pos) * N_HEADS + h) * HEAD_DIM;
                krow = cache_k + co;
                vrow = cache_v + co;
            } else {
                size_t no = (size_t)(b * SEQ + (pos - start_pos)) * D_MODEL + h * HEAD_DIM;
                krow = kb + no;
                vrow = vb + no;
            }
            #pragma unroll
            for (int i = 0; i < 4; i++) {
                float4 fk = *(const float4*)(krow + dp + i * 4);
                float4 fv = *(const float4*)(vrow + dp + i * 4);
                int o = key * 128 + dp + i * 4;
                uint2 pk2; pk2.x = bf2pk(fk.x, fk.y); pk2.y = bf2pk(fk.z, fk.w);
                *(uint2*)&Ksh[o] = pk2;
                int d = dp + i * 4;
                Vt[(d+0)*32+key]=bf1(fv.x); Vt[(d+1)*32+key]=bf1(fv.y);
                Vt[(d+2)*32+key]=bf1(fv.z); Vt[(d+3)*32+key]=bf1(fv.w);
            }
        }
        __syncthreads();

        // ---- scores: 16 q-rows x 32 keys ----
        f32x8 sc[2] = {zero8, zero8};
        const int boff = (lane >> 4) * 8;
        #pragma unroll
        for (int c = 0; c < 4; c++) {
            #pragma unroll
            for (int nt = 0; nt < 2; nt++) {
                FragBF bk_;
                int base = (nt * 16 + (lane & 15)) * 128 + c * 32 + boff;
                bk_.h[0] = *(const ushort8*)&Ksh[base];
                bk_.h[1] = *(const ushort8*)&Ksh[base + 16];
                sc[nt] = __builtin_amdgcn_wmma_f32_16x16x32_bf16(
                    false, aq[c].v, false, bk_.v, (short)0, sc[nt], false, false);
            }
        }
        // dump C-layout scores to LDS rows
        {
            const int rbase = (lane < 16) ? 0 : 8;
            const int col = lane & 15;
            #pragma unroll
            for (int nt = 0; nt < 2; nt++)
                #pragma unroll
                for (int r = 0; r < 8; r++)
                    Ssh[w * 512 + (rbase + r) * 32 + nt * 16 + col] = sc[nt][r];
        }
        __syncthreads();

        // ---- online softmax: lanes 0..15 each own one q-row ----
        if (lane < 16) {
            const float* sr = &Ssh[w * 512 + lane * 32];
            float mx = sr[0];
            #pragma unroll
            for (int j = 1; j < 32; j++) mx = fmaxf(mx, sr[j]);
            float mnew  = fmaxf(mrow, mx);
            float alpha = __expf(mrow - mnew);   // 0 on first chunk
            float acc_s = srow * alpha;
            #pragma unroll
            for (int j = 0; j < 32; j += 2) {
                float p0 = __expf(sr[j]     - mnew);
                float p1 = __expf(sr[j + 1] - mnew);
                acc_s += p0 + p1;
                *(unsigned*)&Psh[w * 512 + lane * 32 + j] = bf2pk(p0, p1);
            }
            Alp[w * 16 + lane] = alpha;
            mrow = mnew;
            srow = acc_s;
        }
        __syncthreads();

        // ---- rescale O and accumulate P x V ----
        {
            const int rbase = (lane < 16) ? 0 : 8;
            float al[8];
            #pragma unroll
            for (int r = 0; r < 8; r++) al[r] = Alp[w * 16 + rbase + r];
            #pragma unroll
            for (int nt = 0; nt < 8; nt++)
                #pragma unroll
                for (int r = 0; r < 8; r++) oacc[nt][r] *= al[r];

            FragBF pf;
            const int aoff = (lane < 16) ? 0 : 8;
            const int pbase = w * 512 + (lane & 15) * 32 + aoff;
            pf.h[0] = *(const ushort8*)&Psh[pbase];
            pf.h[1] = *(const ushort8*)&Psh[pbase + 16];
            const int boff2 = (lane >> 4) * 8;
            #pragma unroll
            for (int nt = 0; nt < 8; nt++) {
                FragBF bv_;
                int base = (nt * 16 + (lane & 15)) * 32 + boff2;
                bv_.h[0] = *(const ushort8*)&Vt[base];
                bv_.h[1] = *(const ushort8*)&Vt[base + 16];
                oacc[nt] = __builtin_amdgcn_wmma_f32_16x16x32_bf16(
                    false, pf.v, false, bv_.v, (short)0, oacc[nt], false, false);
            }
        }
    }

    // ---- normalize and store ----
    if (lane < 16) Sum[w * 16 + lane] = srow;
    __syncthreads();
    {
        const int rbase = (lane < 16) ? 0 : 8;
        const int col = lane & 15;
        float inv[8];
        #pragma unroll
        for (int r = 0; r < 8; r++) inv[r] = 1.f / Sum[w * 16 + rbase + r];
        #pragma unroll
        for (int nt = 0; nt < 8; nt++) {
            #pragma unroll
            for (int r = 0; r < 8; r++) {
                const int q = qblk * 128 + w * 16 + rbase + r;
                out[(size_t)(b * SEQ + q) * D_MODEL + h * HEAD_DIM + nt * 16 + col]
                    = oacc[nt][r] * inv[r];
            }
        }
    }
}

// ---------------------------------------------------------------------------
extern "C" void kernel_launch(void* const* d_in, const int* in_sizes, int n_in,
                              void* d_out, int out_size, void* d_ws, size_t ws_size,
                              hipStream_t stream) {
    (void)in_sizes; (void)n_in; (void)out_size; (void)ws_size;
    const float* x  = (const float*)d_in[0];
    const int*   sp = (const int*)  d_in[1];
    const float* fc = (const float*)d_in[2];
    const float* fs = (const float*)d_in[3];
    const float* wq = (const float*)d_in[4];
    const float* bq = (const float*)d_in[5];
    const float* wk = (const float*)d_in[6];
    const float* bk = (const float*)d_in[7];
    const float* wv = (const float*)d_in[8];
    const float* bv = (const float*)d_in[9];
    const float* ck = (const float*)d_in[10];
    const float* cv = (const float*)d_in[11];
    float* out = (float*)d_out;

    float* qbuf = (float*)d_ws;
    float* kbuf = qbuf + (size_t)BATCH * SEQ * D_MODEL;
    float* vbuf = kbuf + (size_t)BATCH * SEQ * D_MODEL;

    qkv_gemm_kernel<<<dim3(BATCH * SEQ / 128, D_MODEL / 128, 3), 256, 0, stream>>>(
        x, wq, bq, wk, bk, wv, bv, qbuf, kbuf, vbuf);

    rope_kernel<<<dim3((BATCH * SEQ * D_MODEL / 2) / 256, 2, 1), 256, 0, stream>>>(
        qbuf, kbuf, fc, fs);

    attn_kernel<<<dim3(SEQ / 128, N_HEADS, BATCH), 256, 0, stream>>>(
        qbuf, kbuf, vbuf, ck, cv, sp, out);
}
